// DCN_sep_49555332662163
// MI455X (gfx1250) — compile-verified, matrix operand order checked by reading, not run on hardware
//
#include <hip/hip_runtime.h>

// Problem constants (match reference)
#define Bn    2
#define CIN   64
#define COUT  64
#define Hh    224
#define Ww    224
#define KK    9
#define HW    (Hh * Ww)
#define KDIM  (KK * CIN)        // 576
#define KDIMP (KDIM + 8)        // padded LDS row (584 bf16 = 1168 B, 16B aligned)
#define NPIX  (Bn * HW)         // 100352 (divisible by 32)

typedef __bf16 bf16_t;
typedef bf16_t v16bf __attribute__((ext_vector_type(16)));
typedef float  v8f   __attribute__((ext_vector_type(8)));
typedef unsigned v4u __attribute__((ext_vector_type(4)));
typedef int      v4i __attribute__((ext_vector_type(4)));
typedef int      v8i __attribute__((ext_vector_type(8)));

static __device__ __forceinline__ float bf2f(unsigned short u) {
    unsigned v = ((unsigned)u) << 16;
    float f;
    __builtin_memcpy(&f, &v, 4);
    return f;
}
static __device__ __forceinline__ unsigned short f2bf(float f) {
    unsigned v;
    __builtin_memcpy(&v, &f, 4);
    v += 0x7fffu + ((v >> 16) & 1u);   // round-to-nearest-even
    return (unsigned short)(v >> 16);
}

// One 16-element bf16 fragment for wmma_f32_16x16x32_bf16: a lane holds
// K = Kbase+{0..7} and Kbase+16+{0..7} -> two contiguous 16-byte chunks.
static __device__ __forceinline__ v16bf load_frag16(const unsigned short* rowBase, int kOff) {
    union { v16bf v; uint4 q[2]; } u;
    u.q[0] = *(const uint4*)(rowBase + kOff);
    u.q[1] = *(const uint4*)(rowBase + kOff + 16);
    return u.v;
}

// ---------------- Prep: NCHW f32 -> NHWC bf16 ----------------
__global__ void nchw_to_nhwc_bf16(const float* __restrict__ src,
                                  unsigned short* __restrict__ dst, int n) {
    int i = blockIdx.x * blockDim.x + threadIdx.x;
    if (i >= n) return;
    int x = i % Ww; int t = i / Ww;
    int y = t % Hh; t /= Hh;
    int c = t % CIN; int b = t / CIN;
    dst[((size_t)(b * Hh + y) * Ww + x) * CIN + c] = f2bf(src[i]);
}

// ---------------- Prep: weights -> [o][k*64+c] bf16 ----------------
__global__ void prep_woff(const float* __restrict__ w, unsigned short* __restrict__ dst) {
    int i = blockIdx.x * blockDim.x + threadIdx.x;   // 32*576
    if (i >= 32 * KDIM) return;
    int co = i / KDIM; int kc = i % KDIM;
    int k = kc / CIN;  int c = kc % CIN;
    float v = (co < 27) ? w[(co * CIN + c) * KK + k] : 0.f;
    dst[i] = f2bf(v);
}
__global__ void prep_wdcn(const float* __restrict__ w, unsigned short* __restrict__ dst) {
    int i = blockIdx.x * blockDim.x + threadIdx.x;   // 64*576
    if (i >= COUT * KDIM) return;
    int o = i / KDIM; int kc = i % KDIM;
    int k = kc / CIN; int c = kc % CIN;
    dst[i] = f2bf(w[(o * CIN + c) * KK + k]);
}

// ---------------- Offset conv (implicit GEMM, bf16 WMMA + TDM weight stage) ----
// M = 16 pixels, N = 32 (27 padded), K = 576. 2 waves per block.
__global__ __launch_bounds__(64) void offconv_kernel(
    const unsigned short* __restrict__ feaT,   // (B,H,W,C) bf16
    const unsigned short* __restrict__ wToff,  // (32, 576) bf16
    const float* __restrict__ b_off,
    float* __restrict__ om)                    // (B,27,H,W) f32
{
    __shared__ __align__(16) unsigned short As[16 * KDIMP];   // 18688 B im2col
    __shared__ __align__(16) unsigned short Ws[32 * KDIM];    // 36864 B weights (TDM dest)
    __shared__ __align__(16) float          Cs[32 * 17];      //  2176 B C stage

    const int tid  = threadIdx.x;
    const int pix0 = blockIdx.x * 16;
    const int bB   = pix0 / HW;
    const int rp0  = pix0 % HW;

    // --- TDM: stage the whole 576x32 bf16 weight tile Global -> LDS (wave 0) ---
    if (tid < 32) {
        unsigned long long ga = (unsigned long long)(uintptr_t)wToff;
        unsigned ldsa = (unsigned)(uintptr_t)(&Ws[0]);
        v4u g0 = { 1u,                                    // count=1, user mode
                   ldsa,                                  // lds_addr
                   (unsigned)(ga & 0xffffffffu),          // global_addr[31:0]
                   (unsigned)((ga >> 32) & 0x1ffffffu) | (2u << 30) };  // [56:32] | type=2
        v8i g1 = { (int)(1u << 16),          // data_size=2B
                   (int)(576u << 16),        // tensor_dim0 = 576
                   (int)(32u << 16),         // tensor_dim1 = 32
                   (int)(576u << 16),        // tile_dim0   = 576
                   (int)32,                  // tile_dim1   = 32 (tile_dim2=0)
                   (int)576,                 // tensor_dim0_stride = 576
                   (int)(18432u << 16),      // tensor_dim1_stride = 18432 (lo16)
                   0 };
        v4i gz = { 0, 0, 0, 0 };
#if __clang_major__ >= 23
        v8i gz8 = { 0, 0, 0, 0, 0, 0, 0, 0 };
        __builtin_amdgcn_tensor_load_to_lds(g0, g1, gz, gz, gz8, 0);
#else
        __builtin_amdgcn_tensor_load_to_lds(g0, g1, gz, gz, 0);
#endif
        __builtin_amdgcn_s_wait_tensorcnt(0);
    }

    // --- im2col: 16 pixels * 9 taps * 8 chunks(16B) = 1152 -> 18 uniform iters ---
    for (int t = tid; t < 16 * KK * 8; t += 64) {
        int m = t / 72; int r = t % 72;
        int k = r >> 3; int ch = (r & 7) << 3;
        int rp = rp0 + m;
        int y = rp / Ww; int x = rp % Ww;
        int yy = y - 1 + k / 3;
        int xx = x - 1 + k % 3;
        uint4 q = make_uint4(0u, 0u, 0u, 0u);
        if (yy >= 0 && yy < Hh && xx >= 0 && xx < Ww)
            q = *(const uint4*)(feaT + ((size_t)(bB * Hh + yy) * Ww + xx) * CIN + ch);
        *(uint4*)(&As[m * KDIMP + k * CIN + ch]) = q;
    }
    __syncthreads();

    const int lane  = tid & 31;
    const int wave  = tid >> 5;            // 0..1 along N
    const int mrow  = lane & 15;
    const int kbase = (lane >> 4) << 3;
    const int oloc  = wave * 16 + (lane & 15);

    v8f acc = {};
    for (int kb = 0; kb < KDIM / 32; ++kb) {
        v16bf a  = load_frag16(&As[mrow * KDIMP], kb * 32 + kbase);
        v16bf bm = load_frag16(&Ws[oloc * KDIM], kb * 32 + kbase);
        acc = __builtin_amdgcn_wmma_f32_16x16x32_bf16(false, a, false, bm,
                                                      (short)0, acc, false, false);
    }

    // Stage C in LDS: Cs[o][m] so om writes are row-contiguous
#pragma unroll
    for (int r = 0; r < 8; ++r) {
        int m = r + ((lane >> 4) << 3);    // C/D layout: M = r + 8*(lane>=16)
        Cs[oloc * 17 + m] = acc[r];
    }
    __syncthreads();

    // Coalesced om writes: 27 rows * 4 float4 (64B/row) = 108 items
    for (int t = tid; t < 27 * 4; t += 64) {
        int co = t >> 2; int q = (t & 3) << 2;
        float bias = b_off[co];
        float4 v;
        v.x = Cs[co * 17 + q + 0] + bias;
        v.y = Cs[co * 17 + q + 1] + bias;
        v.z = Cs[co * 17 + q + 2] + bias;
        v.w = Cs[co * 17 + q + 3] + bias;
        *(float4*)(om + ((size_t)bB * 27 + co) * HW + rp0 + q) = v;
    }
}

// ---------------- Main DCN (bilinear gather + bf16 WMMA GEMM) ----------------
// M = 32 pixels, N = 64, K = 576. 8 waves per block (2 M-tiles x 4 N-tiles).
__global__ __launch_bounds__(256) void dcn_kernel(
    const unsigned short* __restrict__ inT,   // (B,H,W,C) bf16
    const float* __restrict__ om,             // (B,27,H,W) f32
    const unsigned short* __restrict__ wTd,   // (64, 576) bf16
    const float* __restrict__ b_dcn,
    float* __restrict__ out)                  // (B,64,H,W) f32
{
    __shared__ __align__(16) unsigned short Ss[32 * KDIMP];   // 37376 B sampled matrix
    __shared__ __align__(16) float          Cs[64 * 33];      //  8448 B C stage

    const int tid  = threadIdx.x;
    const int pix0 = blockIdx.x * 32;
    const int bB   = pix0 / HW;
    const int rp0  = pix0 % HW;

    // Build sampled matrix: 32 pixels * 9 taps * 8 chunks = 2304 -> 9 uniform iters
    for (int t = tid; t < 32 * KK * 8; t += 256) {
        int m = t / 72; int r = t % 72;
        int k = r >> 3; int ch = (r & 7) << 3;
        int rp = rp0 + m;
        int y = rp / Ww; int x = rp % Ww;

        const float* omb = om + (size_t)bB * 27 * HW + rp;
        float offy = omb[(size_t)(2 * k) * HW];
        float offx = omb[(size_t)(2 * k + 1) * HW];
        float mv   = omb[(size_t)(18 + k) * HW];
        mv = 1.f / (1.f + __expf(-mv));

        float py = (float)(y - 1 + k / 3) + offy;
        float px = (float)(x - 1 + k % 3) + offx;
        float y0f = floorf(py), x0f = floorf(px);
        int y0 = (int)y0f, x0 = (int)x0f;
        float wy1 = py - y0f, wx1 = px - x0f;
        float w00 = (1.f - wy1) * (1.f - wx1);
        float w01 = (1.f - wy1) * wx1;
        float w10 = wy1 * (1.f - wx1);
        float w11 = wy1 * wx1;

        float acc[8] = {0.f, 0.f, 0.f, 0.f, 0.f, 0.f, 0.f, 0.f};
        const unsigned short* base = inT + (size_t)bB * HW * CIN;

        auto corner = [&](int yy, int xx, float wgt) {
            if (yy >= 0 && yy < Hh && xx >= 0 && xx < Ww) {
                uint4 q = *(const uint4*)(base + ((size_t)yy * Ww + xx) * CIN + ch);
                const unsigned* qd = (const unsigned*)&q;
#pragma unroll
                for (int j = 0; j < 4; ++j) {
                    unsigned v = qd[j];
                    acc[2 * j]     += wgt * bf2f((unsigned short)(v & 0xffffu));
                    acc[2 * j + 1] += wgt * bf2f((unsigned short)(v >> 16));
                }
            }
        };
        corner(y0,     x0,     w00);
        corner(y0,     x0 + 1, w01);
        corner(y0 + 1, x0,     w10);
        corner(y0 + 1, x0 + 1, w11);

        uint4 qo;
        qo.x = (unsigned)f2bf(acc[0] * mv) | ((unsigned)f2bf(acc[1] * mv) << 16);
        qo.y = (unsigned)f2bf(acc[2] * mv) | ((unsigned)f2bf(acc[3] * mv) << 16);
        qo.z = (unsigned)f2bf(acc[4] * mv) | ((unsigned)f2bf(acc[5] * mv) << 16);
        qo.w = (unsigned)f2bf(acc[6] * mv) | ((unsigned)f2bf(acc[7] * mv) << 16);
        *(uint4*)(&Ss[m * KDIMP + k * CIN + ch]) = qo;
    }
    __syncthreads();

    const int lane  = tid & 31;
    const int wave  = tid >> 5;
    const int waveM = wave >> 2;           // 0..1
    const int waveN = wave & 3;            // 0..3
    const int mrow  = waveM * 16 + (lane & 15);
    const int kbase = (lane >> 4) << 3;
    const int ocol  = waveN * 16 + (lane & 15);

    v8f acc = {};
    for (int kb = 0; kb < KDIM / 32; ++kb) {
        v16bf a  = load_frag16(&Ss[mrow * KDIMP], kb * 32 + kbase);
        v16bf bm = load_frag16(wTd + (size_t)ocol * KDIM, kb * 32 + kbase);
        acc = __builtin_amdgcn_wmma_f32_16x16x32_bf16(false, a, false, bm,
                                                      (short)0, acc, false, false);
    }

    // Stage C in LDS: Cs[o][m] so output writes are full 128B lines
#pragma unroll
    for (int r = 0; r < 8; ++r) {
        int m = waveM * 16 + r + ((lane >> 4) << 3);
        Cs[ocol * 33 + m] = acc[r];
    }
    __syncthreads();

    // Coalesced out writes: 64 rows * 8 float4 (128B/row) = 512 items -> 2 iters
    for (int t = tid; t < 64 * 8; t += 256) {
        int o = t >> 3; int q = (t & 7) << 2;
        float bias = b_dcn[o];
        float4 v;
        v.x = Cs[o * 33 + q + 0] + bias;
        v.y = Cs[o * 33 + q + 1] + bias;
        v.z = Cs[o * 33 + q + 2] + bias;
        v.w = Cs[o * 33 + q + 3] + bias;
        *(float4*)(out + ((size_t)bB * COUT + o) * HW + rp0 + q) = v;
    }
}

extern "C" void kernel_launch(void* const* d_in, const int* in_sizes, int n_in,
                              void* d_out, int out_size, void* d_ws, size_t ws_size,
                              hipStream_t stream) {
    (void)in_sizes; (void)n_in; (void)out_size; (void)ws_size;

    const float* input = (const float*)d_in[0];
    const float* fea   = (const float*)d_in[1];
    const float* w_off = (const float*)d_in[2];
    const float* b_off = (const float*)d_in[3];
    const float* w_dcn = (const float*)d_in[4];
    const float* b_dcn = (const float*)d_in[5];
    float* out = (float*)d_out;

    // Workspace layout (all 16B aligned): ~35 MB
    char* ws = (char*)d_ws;
    size_t off = 0;
    float* om = (float*)(ws + off);                     off += (size_t)Bn * 27 * HW * sizeof(float);
    unsigned short* feaT  = (unsigned short*)(ws + off); off += (size_t)Bn * HW * CIN * 2;
    unsigned short* inT   = (unsigned short*)(ws + off); off += (size_t)Bn * HW * CIN * 2;
    unsigned short* wToff = (unsigned short*)(ws + off); off += (size_t)32 * KDIM * 2;
    unsigned short* wTd   = (unsigned short*)(ws + off); off += (size_t)COUT * KDIM * 2;

    const int nT = Bn * CIN * HW;
    nchw_to_nhwc_bf16<<<(nT + 255) / 256, 256, 0, stream>>>(fea, feaT, nT);
    nchw_to_nhwc_bf16<<<(nT + 255) / 256, 256, 0, stream>>>(input, inT, nT);
    prep_woff<<<(32 * KDIM + 255) / 256, 256, 0, stream>>>(w_off, wToff);
    prep_wdcn<<<(COUT * KDIM + 255) / 256, 256, 0, stream>>>(w_dcn, wTd);

    offconv_kernel<<<NPIX / 16, 64, 0, stream>>>(feaT, wToff, b_off, om);
    dcn_kernel<<<NPIX / 32, 256, 0, stream>>>(inT, om, wTd, b_dcn, out);
}
